// DGCNN_MODULAR_10316511445755
// MI455X (gfx1250) — matrix operand order repaired and checked by reading, not compiled, not run on hardware
//
#include <hip/hip_runtime.h>
#include <hip/hip_bf16.h>

#define BB 4
#define NP 1024
#define KNN 20
#define NCH 960
#define PI_F 3.14159265358979323846f

typedef __attribute__((ext_vector_type(16))) _Float16 v16h;
typedef __attribute__((ext_vector_type(8)))  _Float16 v8h;
typedef __attribute__((ext_vector_type(8)))  float    v8f;

// ---------------------------------------------------------------- normalize
__global__ void normalize_kernel(const float* __restrict__ xsrc,
                                 const float* __restrict__ xtgt,
                                 float* __restrict__ outs,
                                 float* __restrict__ outt) {
  int cloud = blockIdx.x / BB;
  int b = blockIdx.x % BB;
  const float* x = (cloud == 0 ? xsrc : xtgt) + (size_t)b * NP * 3;
  float* o = (cloud == 0 ? outs : outt) + (size_t)b * NP * 3;
  __shared__ float red[256];
  __shared__ float mean3[3];
  __shared__ float inv;
  int t = threadIdx.x;
  for (int d = 0; d < 3; ++d) {
    float s = 0.f;
    for (int n = t; n < NP; n += 256) s += x[n * 3 + d];
    red[t] = s; __syncthreads();
    for (int st = 128; st > 0; st >>= 1) {
      if (t < st) red[t] += red[t + st];
      __syncthreads();
    }
    if (t == 0) mean3[d] = red[0] / (float)NP;
    __syncthreads();
  }
  float mx = 0.f;
  for (int n = t; n < NP; n += 256) {
    float a = x[n * 3] - mean3[0];
    float bq = x[n * 3 + 1] - mean3[1];
    float cq = x[n * 3 + 2] - mean3[2];
    mx = fmaxf(mx, sqrtf(a * a + bq * bq + cq * cq));
  }
  red[t] = mx; __syncthreads();
  for (int st = 128; st > 0; st >>= 1) {
    if (t < st) red[t] = fmaxf(red[t], red[t + st]);
    __syncthreads();
  }
  if (t == 0) inv = 1.f / red[0];
  __syncthreads();
  for (int n = t; n < NP; n += 256) {
    o[n * 3]     = (x[n * 3]     - mean3[0]) * inv;
    o[n * 3 + 1] = (x[n * 3 + 1] - mean3[1]) * inv;
    o[n * 3 + 2] = (x[n * 3 + 2] - mean3[2]) * inv;
  }
}

// ---------------------------------------------------------------- OrientNet
__global__ void orient_kernel(const float* __restrict__ xs,
                              const float* __restrict__ xt,
                              const float* __restrict__ Wpt,
                              const float* __restrict__ bpt,
                              const float* __restrict__ Wang,
                              const float* __restrict__ Wd,
                              float* __restrict__ out_orient,
                              float* __restrict__ csb,
                              float* __restrict__ snb) {
  int b = blockIdx.x;
  int t = threadIdx.x;                 // 128 threads
  __shared__ float fs[64], ft[64], ax[12];
  {
    int c = t & 63;
    const float* x = (t < 64 ? xs : xt) + (size_t)b * NP * 3;
    float w0 = Wpt[c * 3], w1 = Wpt[c * 3 + 1], w2 = Wpt[c * 3 + 2], bc = bpt[c];
    float mx = -3.0e38f;
    for (int n = 0; n < NP; ++n) {
      float y = x[n * 3] * w0 + x[n * 3 + 1] * w1 + x[n * 3 + 2] * w2 + bc;
      y = fmaxf(y, 0.f);
      mx = fmaxf(mx, y);
    }
    if (t < 64) fs[c] = mx; else ft[c] = mx;
  }
  __syncthreads();
  if (t < 12) {
    float sx = 0.f, sy = 0.f;
    for (int c = 0; c < 64; ++c) {
      sx += fs[c] * Wang[t * 128 + c] + ft[c] * Wang[t * 128 + 64 + c];
      sy += ft[c] * Wang[t * 128 + c] + fs[c] * Wang[t * 128 + 64 + c];
    }
    ax[t] = sx;
    out_orient[b * 12 + t] = sx;                  // angle_x -> orient_list[0]
    out_orient[BB * 12 + b * 12 + t] = sy;        // angle_y -> orient_list[1]
  }
  if (t == 12) {
    float s = 0.f;
    for (int c = 0; c < 64; ++c) s += fs[c] * Wd[c] + ft[c] * Wd[64 + c];
    out_orient[2 * BB * 12 + b] = 1.f / (1.f + expf(-s));   // global_d
  }
  __syncthreads();
  if (t == 0) {
    int ai = 0; float bv = ax[0];
    for (int i = 1; i < 12; ++i) if (ax[i] > bv) { bv = ax[i]; ai = i; }
    float ANG = -PI_F + (float)ai * (PI_F / 6.f);
    float ang = -ANG;
    csb[b] = cosf(ang);
    snb[b] = sinf(ang);
  }
}

__global__ void rotate_kernel(float* __restrict__ xt,
                              const float* __restrict__ csb,
                              const float* __restrict__ snb) {
  int i = blockIdx.x * blockDim.x + threadIdx.x;
  if (i >= BB * NP) return;
  int b = i / NP;
  float c = csb[b], s = snb[b];
  float* p = xt + (size_t)i * 3;
  float x = p[0], z = p[2];
  p[0] = x * c - z * s;
  p[2] = x * s + z * c;
}

// ---------------------------------------------------------------- converts
// Fh is stored N-MAJOR: Fh[b][n][Kp] so WMMA B/A fragments are contiguous b128 loads.
__global__ void conv_x_to_fh(const float* __restrict__ x, _Float16* __restrict__ Fh) {
  int i = blockIdx.x * 256 + threadIdx.x;
  if (i >= BB * NP * 32) return;
  int c = i % 32;
  int n = (i / 32) % NP;
  int b = i / (32 * NP);
  float v = (c < 3) ? x[((size_t)b * NP + n) * 3 + c] : 0.f;
  Fh[i] = (_Float16)v;
}

__global__ void conv_h_to_fh(const float* __restrict__ H, _Float16* __restrict__ Fh,
                             int coff, int C, int Kp) {
  int i = blockIdx.x * 256 + threadIdx.x;
  if (i >= BB * NP * Kp) return;
  int c = i % Kp;
  int n = (i / Kp) % NP;
  int b = i / (Kp * NP);
  float v = (c < C) ? H[((size_t)b * NCH + coff + c) * NP + n] : 0.f;
  Fh[i] = (_Float16)v;
}

__global__ void wsplit_kernel(const float* __restrict__ W, _Float16* __restrict__ Wnb,
                              _Float16* __restrict__ Wd, int Cout, int C, int Kp) {
  int i = blockIdx.x * 256 + threadIdx.x;
  if (i >= Cout * Kp) return;
  int k = i % Kp, o = i / Kp;
  float a = 0.f, d = 0.f;
  if (k < C) {
    float wnb = W[(size_t)o * 2 * C + k];
    float wct = W[(size_t)o * 2 * C + C + k];
    a = wnb;
    d = wct - wnb;
  }
  Wnb[i] = (_Float16)a;
  Wd[i] = (_Float16)d;
}

__global__ void wconv_kernel(const float* __restrict__ W, _Float16* __restrict__ Wh, int total) {
  int i = blockIdx.x * 256 + threadIdx.x;
  if (i < total) Wh[i] = (_Float16)W[i];
}

__global__ void xx_kernel(const float* __restrict__ H, float* __restrict__ xx,
                          int coff, int C) {
  int i = blockIdx.x * 256 + threadIdx.x;
  if (i >= BB * NP) return;
  int b = i / NP, n = i % NP;
  const float* f = H + ((size_t)b * NCH + coff) * NP + n;
  float s = 0.f;
  for (int c = 0; c < C; ++c) { float v = f[(size_t)c * NP]; s += v * v; }
  xx[i] = s;
}

// ---------------------------------------------------------------- WMMA GEMM
// Out[b][M][N] = Wh[M][Kp] @ Fh[b][N][Kp]^T   (f16 in, f32 out)
// A 16x32 fragment: lane ml=lane&15 is row M, h=lane>>4; e<8 -> k0+8h+e,
// e>=8 -> k0+16+8h+(e-8)  => two contiguous 8-half (16B) loads.
// B 32x16 fragment: lane ml is column N, same k order as A (e<8 -> k0+8h+e,
//   e>=8 -> k0+16+8h+(e-8)) => two contiguous 8-half loads from the n-major Fh row.
__global__ void wmma_gemm_kernel(const _Float16* __restrict__ Wh,
                                 const _Float16* __restrict__ Fh,
                                 float* __restrict__ Out,
                                 int M, int Kp) {
  int lane = threadIdx.x;                       // 0..31
  int wv = threadIdx.y;                         // 0..3
  int tn = blockIdx.x * 16;
  int tm = (blockIdx.y * blockDim.y + wv) * 16;
  if (tm >= M) return;                          // wave-uniform exit
  int b = blockIdx.z;
  int ml = lane & 15;
  int h = lane >> 4;
  const _Float16* Wr = Wh + (size_t)(tm + ml) * Kp;
  const _Float16* Fr = Fh + ((size_t)b * NP + tn + ml) * Kp;
  float* O = Out + (size_t)b * M * NP;
  v8f acc = {};
  for (int k0 = 0; k0 < Kp; k0 += 32) {
    v8h a0 = *(const v8h*)(Wr + k0 + 8 * h);
    v8h a1 = *(const v8h*)(Wr + k0 + 16 + 8 * h);
    v8h b0 = *(const v8h*)(Fr + k0 + 8 * h);
    v8h b1 = *(const v8h*)(Fr + k0 + 16 + 8 * h);
    v16h a = __builtin_shufflevector(a0, a1, 0, 1, 2, 3, 4, 5, 6, 7,
                                     8, 9, 10, 11, 12, 13, 14, 15);
    v16h bf = __builtin_shufflevector(b0, b1, 0, 1, 2, 3, 4, 5, 6, 7,
                                      8, 9, 10, 11, 12, 13, 14, 15);
    acc = __builtin_amdgcn_wmma_f32_16x16x32_f16(false, a, false, bf,
                                                 (short)0, acc, false, false);
  }
#pragma unroll
  for (int v = 0; v < 8; ++v) {
    int m = v + 8 * h;
    O[(size_t)(tm + m) * NP + tn + ml] = acc[v];
  }
}

// Gram: S[b][n][m] = 2 * sum_c F[n][c]*F[m][c] - xx[n] - xx[m]
__global__ void gram_kernel(const _Float16* __restrict__ Fh,
                            const float* __restrict__ xx,
                            float* __restrict__ S, int Kp) {
  int lane = threadIdx.x;
  int wv = threadIdx.y;
  int tn = blockIdx.x * 16;                            // column tile (m)
  int tr = (blockIdx.y * blockDim.y + wv) * 16;        // row tile (n)
  int b = blockIdx.z;
  int ml = lane & 15;
  int h = lane >> 4;
  const _Float16* Ar = Fh + ((size_t)b * NP + tr + ml) * Kp;  // A row = point tr+ml
  const _Float16* Br = Fh + ((size_t)b * NP + tn + ml) * Kp;  // B col = point tn+ml
  v8f acc = {};
  for (int k0 = 0; k0 < Kp; k0 += 32) {
    v8h a0 = *(const v8h*)(Ar + k0 + 8 * h);
    v8h a1 = *(const v8h*)(Ar + k0 + 16 + 8 * h);
    v8h b0 = *(const v8h*)(Br + k0 + 8 * h);
    v8h b1 = *(const v8h*)(Br + k0 + 16 + 8 * h);
    v16h a = __builtin_shufflevector(a0, a1, 0, 1, 2, 3, 4, 5, 6, 7,
                                     8, 9, 10, 11, 12, 13, 14, 15);
    v16h bf = __builtin_shufflevector(b0, b1, 0, 1, 2, 3, 4, 5, 6, 7,
                                      8, 9, 10, 11, 12, 13, 14, 15);
    acc = __builtin_amdgcn_wmma_f32_16x16x32_f16(false, a, false, bf,
                                                 (short)0, acc, false, false);
  }
  const float* xb = xx + (size_t)b * NP;
  float* Sb = S + (size_t)b * NP * NP;
#pragma unroll
  for (int v = 0; v < 8; ++v) {
    int r = tr + v + 8 * h;
    int c = tn + ml;
    Sb[(size_t)r * NP + c] = 2.f * acc[v] - xb[r] - xb[c];
  }
}

// ---------------------------------------------------------------- top-k (k=20)
__global__ void topk_kernel(const float* __restrict__ S, int* __restrict__ idx) {
  int wid = blockIdx.x * (blockDim.x >> 5) + (threadIdx.x >> 5);
  int lane = threadIdx.x & 31;
  if (wid >= BB * NP) return;
  int b = wid / NP, n = wid % NP;
  const float* row = S + ((size_t)b * NP + n) * NP;
  float v[32];
#pragma unroll
  for (int i = 0; i < 32; ++i) v[i] = row[i * 32 + lane];
  int* orow = idx + ((size_t)b * NP + n) * KNN;
  for (int t = 0; t < KNN; ++t) {
    float bv = -3.0e38f;
    int bi = NP;
#pragma unroll
    for (int i = 0; i < 32; ++i) {
      int m = i * 32 + lane;
      if (v[i] > bv) { bv = v[i]; bi = m; }
    }
    for (int off = 16; off > 0; off >>= 1) {
      float ov = __shfl_xor(bv, off, 32);
      int oi = __shfl_xor(bi, off, 32);
      if (ov > bv || (ov == bv && oi < bi)) { bv = ov; bi = oi; }
    }
    if ((bi & 31) == lane) v[bi >> 5] = -3.0e38f;
    if (lane == 0) orow[t] = bi;
  }
}

// ---------------------------------------------------------------- gather + stats
__global__ void gather_stats_kernel(const float* __restrict__ A,
                                    const float* __restrict__ Bt,
                                    const int* __restrict__ idx,
                                    float* __restrict__ MA,
                                    float* __restrict__ ssum,
                                    float* __restrict__ ssq,
                                    int Cout) {
  int o = blockIdx.x, b = blockIdx.y;
  const float* Ab = A + ((size_t)b * Cout + o) * NP;
  const float* Btb = Bt + ((size_t)b * Cout + o) * NP;
  const int* ib = idx + (size_t)b * NP * KNN;
  float* MAb = MA + ((size_t)b * Cout + o) * NP;
  float lsum = 0.f, lsq = 0.f;
  for (int n = threadIdx.x; n < NP; n += 256) {
    if (n + 256 < NP) __builtin_prefetch(ib + (n + 256) * KNN, 0, 1);
    float bt = Btb[n];
    float sa = 0.f, sa2 = 0.f, amax = -3.0e38f;
#pragma unroll
    for (int k = 0; k < KNN; ++k) {
      int j = ib[n * KNN + k];
      float a = Ab[j];
      sa += a; sa2 += a * a;
      amax = fmaxf(amax, a);
    }
    lsum += sa + (float)KNN * bt;
    lsq += sa2 + 2.f * bt * sa + (float)KNN * bt * bt;
    MAb[n] = amax;
  }
  __shared__ float r1[256], r2[256];
  r1[threadIdx.x] = lsum;
  r2[threadIdx.x] = lsq;
  __syncthreads();
  for (int st = 128; st > 0; st >>= 1) {
    if (threadIdx.x < st) {
      r1[threadIdx.x] += r1[threadIdx.x + st];
      r2[threadIdx.x] += r2[threadIdx.x + st];
    }
    __syncthreads();
  }
  if (threadIdx.x == 0) {
    atomicAdd(&ssum[o], r1[0]);
    atomicAdd(&ssq[o], r2[0]);
  }
}

__global__ void zero_kernel(float* __restrict__ p, int n) {
  int i = blockIdx.x * 256 + threadIdx.x;
  if (i < n) p[i] = 0.f;
}

__global__ void bn_final_kernel(const float* __restrict__ ssum, const float* __restrict__ ssq,
                                const float* __restrict__ g, const float* __restrict__ bb,
                                float* __restrict__ sc, float* __restrict__ sh,
                                int Cout, float invcnt) {
  int o = blockIdx.x * blockDim.x + threadIdx.x;
  if (o >= Cout) return;
  float m = ssum[o] * invcnt;
  float v = ssq[o] * invcnt - m * m;
  float s = g[o] * rsqrtf(v + 1e-5f);
  sc[o] = s;
  sh[o] = bb[o] - m * s;
}

__global__ void apply_edge_kernel(const float* __restrict__ MA, const float* __restrict__ Bt,
                                  const float* __restrict__ sc, const float* __restrict__ sh,
                                  float* __restrict__ H, int Cout, int coff) {
  int i = blockIdx.x * 256 + threadIdx.x;
  if (i >= BB * Cout * NP) return;
  int n = i % NP;
  int o = (i / NP) % Cout;
  int b = i / (NP * Cout);
  float y = sc[o] * (MA[i] + Bt[i]) + sh[o];
  H[((size_t)b * NCH + coff + o) * NP + n] = (y >= 0.f) ? y : 0.2f * y;
}

__global__ void stats_final_kernel(const float* __restrict__ Y,
                                   float* __restrict__ ssum, float* __restrict__ ssq) {
  int o = blockIdx.x, b = blockIdx.y;
  const float* yb = Y + ((size_t)b * 512 + o) * NP;
  float ls = 0.f, lq = 0.f;
  for (int n = threadIdx.x; n < NP; n += 256) {
    float v = yb[n];
    ls += v; lq += v * v;
  }
  __shared__ float r1[256], r2[256];
  r1[threadIdx.x] = ls;
  r2[threadIdx.x] = lq;
  __syncthreads();
  for (int st = 128; st > 0; st >>= 1) {
    if (threadIdx.x < st) {
      r1[threadIdx.x] += r1[threadIdx.x + st];
      r2[threadIdx.x] += r2[threadIdx.x + st];
    }
    __syncthreads();
  }
  if (threadIdx.x == 0) {
    atomicAdd(&ssum[o], ls = r1[0]);
    atomicAdd(&ssq[o], r2[0]);
  }
}

__global__ void apply_final_kernel(const float* __restrict__ Y,
                                   const float* __restrict__ sc, const float* __restrict__ sh,
                                   float* __restrict__ out) {
  int i = blockIdx.x * 256 + threadIdx.x;
  if (i >= BB * 512 * NP) return;
  int o = (i / NP) % 512;
  float y = sc[o] * Y[i] + sh[o];
  out[i] = (y >= 0.f) ? y : 0.2f * y;
}

// ---------------------------------------------------------------- launch
extern "C" void kernel_launch(void* const* d_in, const int* in_sizes, int n_in,
                              void* d_out, int out_size, void* d_ws, size_t ws_size,
                              hipStream_t stream) {
  const float* x_src = (const float*)d_in[0];
  const float* x_tgt = (const float*)d_in[1];
  const int* sn_src = (const int*)d_in[2];
  const int* sn_tgt = (const int*)d_in[3];
  const float* Wl = (const float*)d_in[16];
  const float* gl = (const float*)d_in[17];
  const float* bl = (const float*)d_in[18];
  const float* Wpt = (const float*)d_in[19];
  const float* bpt = (const float*)d_in[20];
  const float* Wang = (const float*)d_in[21];
  const float* Wd = (const float*)d_in[22];
  float* out = (float*)d_out;

  // workspace carve-up (256B aligned)
  char* wp = (char*)d_ws;
  auto alloc = [&](size_t bytes) -> void* {
    void* p = (void*)wp;
    wp += (bytes + 255) & ~(size_t)255;
    return p;
  };
  float* xs   = (float*)alloc((size_t)BB * NP * 3 * 4);
  float* xt   = (float*)alloc((size_t)BB * NP * 3 * 4);
  float* csb  = (float*)alloc(BB * 4);
  float* snb  = (float*)alloc(BB * 4);
  float* H    = (float*)alloc((size_t)BB * NCH * NP * 4);
  float* Abuf = (float*)alloc((size_t)BB * 512 * NP * 4);
  float* Btbf = (float*)alloc((size_t)BB * 512 * NP * 4);
  float* MAbf = (float*)alloc((size_t)BB * 512 * NP * 4);
  float* Sm   = (float*)alloc((size_t)BB * NP * NP * 4);
  float* xx   = (float*)alloc((size_t)BB * NP * 4);
  float* stats = (float*)alloc(1024 * 4);          // ssum[512] | ssq[512]
  float* ssum = stats, * ssq = stats + 512;
  float* scale = (float*)alloc(1024 * 4);          // sc[512] | sh[512]
  float* sc = scale, * sh = scale + 512;
  int* idxb = (int*)alloc((size_t)BB * NP * KNN * 4);
  _Float16* Fh   = (_Float16*)alloc((size_t)BB * NP * NCH * 2);
  _Float16* Wnbh = (_Float16*)alloc((size_t)512 * NCH * 2);
  _Float16* Wdh  = (_Float16*)alloc((size_t)512 * NCH * 2);

  // stage A/B/C: normalize, orient heads, rotate target
  normalize_kernel<<<2 * BB, 256, 0, stream>>>(x_src, x_tgt, xs, xt);
  orient_kernel<<<BB, 128, 0, stream>>>(xs, xt, Wpt, bpt, Wang, Wd,
                                        out + (size_t)2 * BB * 512 * NP, csb, snb);
  rotate_kernel<<<(BB * NP + 255) / 256, 256, 0, stream>>>(xt, csb, snb);

  struct LC { int C, Cout, coff, Kp; const float *W, *g, *bb; };
  LC L[4] = {
    {3,   64,  0,   32,  (const float*)d_in[4],  (const float*)d_in[5],  (const float*)d_in[6]},
    {64,  128, 64,  64,  (const float*)d_in[7],  (const float*)d_in[8],  (const float*)d_in[9]},
    {128, 256, 192, 128, (const float*)d_in[10], (const float*)d_in[11], (const float*)d_in[12]},
    {256, 512, 448, 256, (const float*)d_in[13], (const float*)d_in[14], (const float*)d_in[15]},
  };

  for (int br = 0; br < 2; ++br) {
    const float* X = br ? xt : xs;
    const int* sidx = br ? sn_tgt : sn_src;
    float* dst = out + (size_t)br * BB * 512 * NP;

    for (int li = 0; li < 4; ++li) {
      const LC& lc = L[li];
      const int* idxp;
      if (li == 0) {
        conv_x_to_fh<<<(BB * NP * 32 + 255) / 256, 256, 0, stream>>>(X, Fh);
        idxp = sidx;
      } else {
        int pcoff = L[li - 1].coff;
        conv_h_to_fh<<<(BB * NP * lc.Kp + 255) / 256, 256, 0, stream>>>(H, Fh, pcoff, lc.C, lc.Kp);
        xx_kernel<<<(BB * NP + 255) / 256, 256, 0, stream>>>(H, xx, pcoff, lc.C);
        gram_kernel<<<dim3(NP / 16, NP / 16 / 4, BB), dim3(32, 4), 0, stream>>>(Fh, xx, Sm, lc.Kp);
        topk_kernel<<<(BB * NP + 7) / 8, 256, 0, stream>>>(Sm, idxb);
        idxp = idxb;
      }
      wsplit_kernel<<<(lc.Cout * lc.Kp + 255) / 256, 256, 0, stream>>>(lc.W, Wnbh, Wdh,
                                                                       lc.Cout, lc.C, lc.Kp);
      wmma_gemm_kernel<<<dim3(NP / 16, (lc.Cout + 63) / 64, BB), dim3(32, 4), 0, stream>>>(
          Wnbh, Fh, Abuf, lc.Cout, lc.Kp);
      wmma_gemm_kernel<<<dim3(NP / 16, (lc.Cout + 63) / 64, BB), dim3(32, 4), 0, stream>>>(
          Wdh, Fh, Btbf, lc.Cout, lc.Kp);
      zero_kernel<<<4, 256, 0, stream>>>(stats, 1024);
      gather_stats_kernel<<<dim3(lc.Cout, BB), 256, 0, stream>>>(Abuf, Btbf, idxp, MAbf,
                                                                 ssum, ssq, lc.Cout);
      bn_final_kernel<<<(lc.Cout + 63) / 64, 64, 0, stream>>>(
          ssum, ssq, lc.g, lc.bb, sc, sh, lc.Cout, 1.f / (float)(BB * NP * KNN));
      apply_edge_kernel<<<(BB * lc.Cout * NP + 255) / 256, 256, 0, stream>>>(
          MAbf, Btbf, sc, sh, H, lc.Cout, lc.coff);
    }

    // final 960 -> 512 conv + BN(axes 0,2) + LReLU
    conv_h_to_fh<<<(BB * NP * NCH + 255) / 256, 256, 0, stream>>>(H, Fh, 0, NCH, NCH);
    wconv_kernel<<<(512 * NCH + 255) / 256, 256, 0, stream>>>(Wl, Wnbh, 512 * NCH);
    wmma_gemm_kernel<<<dim3(NP / 16, 8, BB), dim3(32, 4), 0, stream>>>(Wnbh, Fh, Abuf, 512, NCH);
    zero_kernel<<<4, 256, 0, stream>>>(stats, 1024);
    stats_final_kernel<<<dim3(512, BB), 256, 0, stream>>>(Abuf, ssum, ssq);
    bn_final_kernel<<<8, 64, 0, stream>>>(ssum, ssq, gl, bl, sc, sh, 512,
                                          1.f / (float)(BB * NP));
    apply_final_kernel<<<(BB * 512 * NP + 255) / 256, 256, 0, stream>>>(Abuf, sc, sh, dst);
  }
}
